// Net_SLSTM_56272661512582
// MI455X (gfx1250) — matrix operand order repaired and checked by reading, not compiled, not run on hardware
//
#include <hip/hip_runtime.h>

// ---------------- problem constants ----------------
#define T_STEPS 400
#define BATCH   256
#define HID     512
#define GATES   2048              // 4*HID
#define XPADK   32                // K=14 input padded to 32 for WMMA
#define NWG     64                // persistent grid: 64 workgroups
#define BLOCK   256               // 8 waves (wave32)
#define NTHREADS (NWG * BLOCK)    // 16384

typedef __bf16 bf16;
typedef __attribute__((ext_vector_type(16))) __bf16 v16bf;
typedef __attribute__((ext_vector_type(8)))  __bf16 v8bf;
typedef __attribute__((ext_vector_type(8)))  float  v8f;

// ---------------- CDNA5 async copy: global -> LDS (ASYNCcnt tracked) -------
__device__ __forceinline__ void async_b128(unsigned lds_off, const void* g) {
  asm volatile("global_load_async_to_lds_b128 %0, %1, off"
               :: "v"(lds_off), "v"(g) : "memory");
}
__device__ __forceinline__ void wait_async0() {
  asm volatile("s_wait_asynccnt 0x0" ::: "memory");
}

// Stage `bytes` contiguous bytes into LDS; 16B per thread per iteration.
__device__ __forceinline__ void stage_panel(unsigned lds_base, const char* g, int bytes) {
  for (int off = threadIdx.x * 16; off < bytes; off += BLOCK * 16)
    async_b128(lds_base + (unsigned)off, g + off);
}

// ---------------- WMMA fragment loads (per CDNA5 ISA layouts) ----
// A 16x32 bf16: lane = {hf = lane>>4, ml = lane&15}; lane holds
// K = kb + 16*hf + [0..15] contiguous (matches B fragment order).
__device__ __forceinline__ v16bf load_a_frag(const bf16* act, int row, int ldk,
                                             int kb, int hf) {
  const bf16* p = act + row * ldk + kb + 16 * hf;
  v8bf lo = *(const v8bf*)p;          // 16B aligned (ds_load_b128 when LDS)
  v8bf hi = *(const v8bf*)(p + 8);
  v16bf a;
#pragma unroll
  for (int i = 0; i < 8; ++i) { a[i] = lo[i]; a[i + 8] = hi[i]; }
  return a;
}

// B 32x16 bf16 (B[k][n] = W[n][k], W row-major): lane n = lane&15,
// khalf = lane>>4; lane holds K = kb + 16*khalf + [0..15] contiguous.
__device__ __forceinline__ v16bf load_b_frag(const bf16* __restrict__ w,
                                             int row, int ldk, int kb, int hf) {
  return *(const v16bf*)(w + row * ldk + kb + 16 * hf);  // 32B aligned
}

// C/D 16x16 f32: VGPR v holds M = v + 8*(lane>>4), N = lane&15.
__device__ __forceinline__ void store_c(float* __restrict__ out,
                                        int m0, int n0, const v8f& c,
                                        int ml, int hf) {
#pragma unroll
  for (int v = 0; v < 8; ++v)
    out[(m0 + v + 8 * hf) * GATES + n0 + ml] = c[v];
}

// 32x32 wave tile, A panel in LDS (rows 0..31 local), B from global (L2).
// unroll capped at 2 so B-fragment loads stay loop-variant (no LICM mass
// hoist across the 400-step t-loop -> no scratch spills).
__device__ __forceinline__ void mma_block_lds(v8f c[4], const bf16* sA,
                                              const bf16* __restrict__ w,
                                              int n0, int K, int ml, int hf) {
#pragma unroll 2
  for (int kb = 0; kb < K; kb += 32) {
    v16bf a0 = load_a_frag(sA, ml,      K, kb, hf);
    v16bf a1 = load_a_frag(sA, 16 + ml, K, kb, hf);
    v16bf b0 = load_b_frag(w, n0 + ml,      K, kb, hf);
    v16bf b1 = load_b_frag(w, n0 + 16 + ml, K, kb, hf);
    c[0] = __builtin_amdgcn_wmma_f32_16x16x32_bf16(false, a0, false, b0, (short)0, c[0], false, false);
    c[1] = __builtin_amdgcn_wmma_f32_16x16x32_bf16(false, a0, false, b1, (short)0, c[1], false, false);
    c[2] = __builtin_amdgcn_wmma_f32_16x16x32_bf16(false, a1, false, b0, (short)0, c[2], false, false);
    c[3] = __builtin_amdgcn_wmma_f32_16x16x32_bf16(false, a1, false, b1, (short)0, c[3], false, false);
  }
}

// ---------------- grid-wide barrier (device scope, sense-reversing) --------
__device__ __forceinline__ void grid_sync(unsigned* bar) {
  __syncthreads();
  if (threadIdx.x == 0) {
    __threadfence();  // release my stores to device scope
    unsigned gen = __hip_atomic_load(bar + 1, __ATOMIC_RELAXED, __HIP_MEMORY_SCOPE_AGENT);
    unsigned arrived = __hip_atomic_fetch_add(bar, 1u, __ATOMIC_ACQ_REL, __HIP_MEMORY_SCOPE_AGENT);
    if (arrived == NWG - 1) {
      __hip_atomic_store(bar, 0u, __ATOMIC_RELAXED, __HIP_MEMORY_SCOPE_AGENT);
      __hip_atomic_fetch_add(bar + 1, 1u, __ATOMIC_RELEASE, __HIP_MEMORY_SCOPE_AGENT);
    } else {
      while (__hip_atomic_load(bar + 1, __ATOMIC_ACQUIRE, __HIP_MEMORY_SCOPE_AGENT) == gen)
        __builtin_amdgcn_s_sleep(2);
    }
  }
  __syncthreads();
}

__device__ __forceinline__ float sigf(float x) { return 1.0f / (1.0f + __expf(-x)); }

// ---------------- prologue: f32->bf16 weight/x conversion, bias fusion -----
__global__ void prep_kernel(const float* __restrict__ x,
                            const float* __restrict__ Wih1,
                            const float* __restrict__ Whh1,
                            const float* __restrict__ bih1, const float* __restrict__ bhh1,
                            const float* __restrict__ Wih2,
                            const float* __restrict__ Whh2,
                            const float* __restrict__ bih2, const float* __restrict__ bhh2,
                            bf16* Whh1b, bf16* Wih2b, bf16* Whh2b,
                            bf16* Wih1p, float* bias1, float* bias2, bf16* xpad) {
  int tid = blockIdx.x * blockDim.x + threadIdx.x;
  int nt  = gridDim.x * blockDim.x;
  for (int i = tid; i < GATES * HID; i += nt) {
    Whh1b[i] = (bf16)Whh1[i];
    Wih2b[i] = (bf16)Wih2[i];
    Whh2b[i] = (bf16)Whh2[i];
  }
  for (int i = tid; i < GATES * XPADK; i += nt) {
    int n = i >> 5, k = i & 31;
    Wih1p[i] = (bf16)((k < 14) ? Wih1[n * 14 + k] : 0.0f);
  }
  for (int i = tid; i < GATES; i += nt) {
    bias1[i] = bih1[i] + bhh1[i];
    bias2[i] = bih2[i] + bhh2[i];
  }
  for (int i = tid; i < T_STEPS * BATCH * XPADK; i += nt) {
    int r = i >> 5, k = i & 31;
    xpad[i] = (bf16)((k < 14) ? x[r * 14 + k] : 0.0f);
  }
}

// ---------------- persistent recurrent kernel ------------------------------
// Dynamic LDS layout: [panel0 32KB][panel1 32KB][x panel 2KB] = 67584 B
#define LDS_BYTES (2 * 32 * HID * 2 + 32 * XPADK * 2)

__global__ void __launch_bounds__(BLOCK, 1)
slstm_persistent(const bf16* __restrict__ xpad,
                 const bf16* __restrict__ Wih1p, const bf16* __restrict__ Whh1b,
                 const bf16* __restrict__ Wih2b, const bf16* __restrict__ Whh2b,
                 const float* __restrict__ bias1, const float* __restrict__ bias2,
                 float* gates1, float* gates2,
                 float* syn1, float* mem1, float* syn2, float* mem2, float* mem2_sum,
                 bf16* mem1_bf, bf16* spk1_bf, bf16* mem2_bf,
                 const float* thr1p, const float* thr2p,
                 unsigned* bar) {
  extern __shared__ bf16 smem[];
  bf16* sP0 = smem;                 // 32 x 512 bf16
  bf16* sP1 = smem + 32 * HID;      // 32 x 512 bf16
  bf16* sX  = smem + 64 * HID;      // 32 x 32 bf16
  const unsigned ldsP0 = (unsigned)(uintptr_t)(void*)sP0;
  const unsigned ldsP1 = (unsigned)(uintptr_t)(void*)sP1;
  const unsigned ldsX  = (unsigned)(uintptr_t)(void*)sX;

  const int lane  = threadIdx.x & 31;
  const int wid   = threadIdx.x >> 5;
  const int gwave = blockIdx.x * (BLOCK / 32) + wid;  // 0..511
  const int ml = lane & 15;
  const int hf = lane >> 4;
  const int m0 = (gwave >> 6) * 32;   // 8 M-blocks of 32 rows (same for all
                                      // 8 waves of a WG -> shared A panel)
  const int n0 = (gwave & 63) * 32;   // 64 N-blocks of 32 cols
  const float thr1 = *thr1p;
  const float thr2 = *thr2p;
  const int tid = blockIdx.x * BLOCK + threadIdx.x;   // 0..16383

  for (int t = 0; t < T_STEPS; ++t) {
    // ---- Phase A: gates1 = bias1 + x_t @ Wih1^T + mem1 @ Whh1^T
    {
      // stage shared A panels into LDS via async copies
      stage_panel(ldsX,  (const char*)(xpad + ((size_t)t * BATCH + m0) * XPADK),
                  32 * XPADK * 2);
      stage_panel(ldsP0, (const char*)(mem1_bf + (size_t)m0 * HID), 32 * HID * 2);
      wait_async0();
      __syncthreads();

      v8f c[4];
      float b0 = bias1[n0 + ml];
      float b1 = bias1[n0 + 16 + ml];
#pragma unroll
      for (int v = 0; v < 8; ++v) { c[0][v] = b0; c[1][v] = b1; c[2][v] = b0; c[3][v] = b1; }
      mma_block_lds(c, sX,  Wih1p, n0, XPADK, ml, hf);
      mma_block_lds(c, sP0, Whh1b, n0, HID,   ml, hf);
      store_c(gates1, m0,      n0,      c[0], ml, hf);
      store_c(gates1, m0,      n0 + 16, c[1], ml, hf);
      store_c(gates1, m0 + 16, n0,      c[2], ml, hf);
      store_c(gates1, m0 + 16, n0 + 16, c[3], ml, hf);
    }
    grid_sync(bar);

    // ---- Phase B: layer-1 elementwise LSTM/spike update
#pragma unroll
    for (int i = 0; i < 8; ++i) {
      int idx = tid + i * NTHREADS;
      int b = idx >> 9, h = idx & (HID - 1);
      const float* g = gates1 + b * GATES;
      float ig = g[h], fg = g[HID + h], gg = g[2 * HID + h], og = g[3 * HID + h];
      float syn = syn1[idx], mo = mem1[idx];
      float reset = (mo - thr1) > 0.0f ? thr1 : 0.0f;
      float sn = sigf(fg) * syn + sigf(ig) * tanhf(gg);
      float hc = sigf(og) * tanhf(sn);
      float mn = hc - reset;
      float spk = (mn - thr1) > 0.0f ? 1.0f : 0.0f;
      syn1[idx] = sn; mem1[idx] = mn;
      mem1_bf[idx] = (bf16)mn; spk1_bf[idx] = (bf16)spk;
    }
    // prefetch next timestep's x panel into cache (global_prefetch_b8)
    if (t + 1 < T_STEPS && threadIdx.x < 32)
      __builtin_prefetch(xpad + ((size_t)(t + 1) * BATCH + m0) * XPADK +
                         (size_t)threadIdx.x * 32, 0, 3);
    grid_sync(bar);

    // ---- Phase C: gates2 = bias2 + spk1 @ Wih2^T + mem2 @ Whh2^T
    {
      stage_panel(ldsP0, (const char*)(spk1_bf + (size_t)m0 * HID), 32 * HID * 2);
      stage_panel(ldsP1, (const char*)(mem2_bf + (size_t)m0 * HID), 32 * HID * 2);
      wait_async0();
      __syncthreads();

      v8f c[4];
      float b0 = bias2[n0 + ml];
      float b1 = bias2[n0 + 16 + ml];
#pragma unroll
      for (int v = 0; v < 8; ++v) { c[0][v] = b0; c[1][v] = b1; c[2][v] = b0; c[3][v] = b1; }
      mma_block_lds(c, sP0, Wih2b, n0, HID, ml, hf);
      mma_block_lds(c, sP1, Whh2b, n0, HID, ml, hf);
      store_c(gates2, m0,      n0,      c[0], ml, hf);
      store_c(gates2, m0,      n0 + 16, c[1], ml, hf);
      store_c(gates2, m0 + 16, n0,      c[2], ml, hf);
      store_c(gates2, m0 + 16, n0 + 16, c[3], ml, hf);
    }
    grid_sync(bar);

    // ---- Phase D: layer-2 elementwise + running mean accumulation
#pragma unroll
    for (int i = 0; i < 8; ++i) {
      int idx = tid + i * NTHREADS;
      int b = idx >> 9, h = idx & (HID - 1);
      const float* g = gates2 + b * GATES;
      float ig = g[h], fg = g[HID + h], gg = g[2 * HID + h], og = g[3 * HID + h];
      float syn = syn2[idx], mo = mem2[idx];
      float reset = (mo - thr2) > 0.0f ? thr2 : 0.0f;
      float sn = sigf(fg) * syn + sigf(ig) * tanhf(gg);
      float hc = sigf(og) * tanhf(sn);
      float mn = hc - reset;
      syn2[idx] = sn; mem2[idx] = mn;
      mem2_bf[idx] = (bf16)mn;
      mem2_sum[idx] += mn;
    }
    grid_sync(bar);
  }
}

// ---------------- epilogue: out = (mem2_sum/T) @ Wfc^T + bfc ---------------
__global__ void final_fc(const float* __restrict__ mem2_sum,
                         const float* __restrict__ Wfc,
                         const float* __restrict__ bfc,
                         float* __restrict__ out) {
  int tid = blockIdx.x * blockDim.x + threadIdx.x;
  if (tid >= BATCH * 8) return;
  int b = tid >> 3, c = tid & 7;
  float acc = 0.0f;
  for (int h = 0; h < HID; ++h)
    acc += mem2_sum[b * HID + h] * Wfc[c * HID + h];
  out[tid] = acc * (1.0f / (float)T_STEPS) + bfc[c];
}

// ---------------- host launcher --------------------------------------------
extern "C" void kernel_launch(void* const* d_in, const int* in_sizes, int n_in,
                              void* d_out, int out_size, void* d_ws, size_t ws_size,
                              hipStream_t stream) {
  (void)in_sizes; (void)n_in; (void)out_size; (void)ws_size;
  const float* x     = (const float*)d_in[0];
  const float* Wih1  = (const float*)d_in[1];
  const float* Whh1  = (const float*)d_in[2];
  const float* bih1  = (const float*)d_in[3];
  const float* bhh1  = (const float*)d_in[4];
  const float* thr1  = (const float*)d_in[5];
  const float* Wih2  = (const float*)d_in[6];
  const float* Whh2  = (const float*)d_in[7];
  const float* bih2  = (const float*)d_in[8];
  const float* bhh2  = (const float*)d_in[9];
  const float* thr2  = (const float*)d_in[10];
  const float* Wfc   = (const float*)d_in[11];
  const float* bfc   = (const float*)d_in[12];
  float* out = (float*)d_out;

  // workspace layout (all chunk sizes are multiples of 256B)
  char* ws = (char*)d_ws;
  size_t off = 0;
  unsigned* bar = (unsigned*)(ws + off); off += 256;
  const size_t BHf = (size_t)BATCH * HID * sizeof(float);   // 512 KB
  const size_t BHb = (size_t)BATCH * HID * sizeof(bf16);    // 256 KB
  float* syn1     = (float*)(ws + off); off += BHf;
  float* mem1     = (float*)(ws + off); off += BHf;
  float* syn2     = (float*)(ws + off); off += BHf;
  float* mem2     = (float*)(ws + off); off += BHf;
  float* mem2_sum = (float*)(ws + off); off += BHf;
  bf16* mem1_bf   = (bf16*)(ws + off); off += BHb;
  bf16* spk1_bf   = (bf16*)(ws + off); off += BHb;
  bf16* mem2_bf   = (bf16*)(ws + off); off += BHb;
  const size_t zero_bytes = off;  // barrier + all recurrent state
  float* gates1 = (float*)(ws + off); off += (size_t)BATCH * GATES * sizeof(float);
  float* gates2 = (float*)(ws + off); off += (size_t)BATCH * GATES * sizeof(float);
  bf16* Whh1b = (bf16*)(ws + off); off += (size_t)GATES * HID * sizeof(bf16);
  bf16* Wih2b = (bf16*)(ws + off); off += (size_t)GATES * HID * sizeof(bf16);
  bf16* Whh2b = (bf16*)(ws + off); off += (size_t)GATES * HID * sizeof(bf16);
  bf16* Wih1p = (bf16*)(ws + off); off += (size_t)GATES * XPADK * sizeof(bf16);
  float* bias1 = (float*)(ws + off); off += (size_t)GATES * sizeof(float);
  float* bias2 = (float*)(ws + off); off += (size_t)GATES * sizeof(float);
  bf16* xpad  = (bf16*)(ws + off); off += (size_t)T_STEPS * BATCH * XPADK * sizeof(bf16);

  hipMemsetAsync(d_ws, 0, zero_bytes, stream);
  prep_kernel<<<512, 256, 0, stream>>>(x, Wih1, Whh1, bih1, bhh1, Wih2, Whh2,
                                       bih2, bhh2, Whh1b, Wih2b, Whh2b,
                                       Wih1p, bias1, bias2, xpad);
  slstm_persistent<<<NWG, BLOCK, LDS_BYTES, stream>>>(
      xpad, Wih1p, Whh1b, Wih2b, Whh2b, bias1, bias2, gates1, gates2,
      syn1, mem1, syn2, mem2, mem2_sum, mem1_bf, spk1_bf, mem2_bf,
      thr1, thr2, bar);
  final_fc<<<8, 256, 0, stream>>>(mem2_sum, Wfc, bfc, out);
}
